// GlycanTreeEncoder_38259568673205
// MI455X (gfx1250) — matrix operand order repaired and checked by reading, not compiled, not run on hardware
//
#include <hip/hip_runtime.h>

typedef _Float16 f16;
typedef _Float16 v16h __attribute__((ext_vector_type(16)));
typedef _Float16 h8   __attribute__((ext_vector_type(8)));
typedef float    v8f  __attribute__((ext_vector_type(8)));

#define NN 32256   // nodes = 512*63
#define EE 31744   // edges = 512*62
#define GG 512     // graphs
#define LDS_STRIDE 40

// ---------------------------------------------------------------------------
// WMMA GEMM: C[M,Nc] = act(A[M,K](f16) @ B[K,Nc](f16) + bias)
// B is supplied TRANSPOSED: BT[Nc][ldbt] (weights are pre-formatted on device),
// so both A and B LDS staging are straight b128 copies (no in-kernel transpose).
// Block tile 128x128, 8 waves (wave32), wave tile 64x32 = 4x2 WMMA 16x16x32.
// Optionally writes f32 C (ldc) and/or f16 C16 (ldc16, for chained GEMMs /
// in-place concatenation). Requires K%32==0, Nc%128==0; M is guarded.
// act: 0=none, 1=GELU(erf), 2=tanh
// ---------------------------------------------------------------------------
__global__ __launch_bounds__(256)
void wmma_gemm(const f16* __restrict__ A, int lda,
               const f16* __restrict__ BT, int ldbt,
               const float* __restrict__ bias,
               float* __restrict__ C, int ldc,
               f16* __restrict__ C16, int ldc16,
               int M, int Nc, int K, int act)
{
    __shared__ __align__(16) f16 As[128 * LDS_STRIDE];
    __shared__ __align__(16) f16 Bs[128 * LDS_STRIDE];   // Bs[col][k]

    const int tid  = threadIdx.x;
    const int lane = tid & 31;
    const int wave = tid >> 5;
    const int wm   = wave >> 2;        // 0..1  (64-row slab)
    const int wn   = wave & 3;         // 0..3  (32-col slab)
    const int m0   = blockIdx.y * 128;
    const int n0   = blockIdx.x * 128;

    v8f acc[4][2];
    for (int i = 0; i < 4; i++)
        for (int j = 0; j < 2; j++)
            for (int v = 0; v < 8; v++) acc[i][j][v] = 0.0f;

    const int srow = tid >> 1;          // 0..127 (A row / B col)
    const int skof = (tid & 1) * 16;    // 0 | 16

    for (int kt = 0; kt < K; kt += 32) {
        __syncthreads();
        // --- stage A tile 128x32 ---
        {
            int gr = m0 + srow;
            h8 v0, v1;
            if (gr < M) {
                const h8* p = (const h8*)(A + (size_t)gr * lda + kt + skof);
                v0 = p[0]; v1 = p[1];
            } else {
                for (int j = 0; j < 8; j++) { v0[j] = (f16)0.0f; v1[j] = (f16)0.0f; }
            }
            f16* d = As + srow * LDS_STRIDE + skof;
            *(h8*)(d)     = v0;
            *(h8*)(d + 8) = v1;
        }
        // --- stage B tile (128 cols x 32 k) from BT, straight copy ---
        {
            const h8* p = (const h8*)(BT + (size_t)(n0 + srow) * ldbt + kt + skof);
            h8 v0 = p[0], v1 = p[1];
            f16* d = Bs + srow * LDS_STRIDE + skof;
            *(h8*)(d)     = v0;
            *(h8*)(d + 8) = v1;
        }
        __syncthreads();

        // prefetch next K tile while compute runs
        if (kt + 32 < K) {
            int gr = m0 + srow;
            if (gr < M) __builtin_prefetch(A + (size_t)gr * lda + kt + 32 + skof, 0, 1);
            __builtin_prefetch(BT + (size_t)(n0 + srow) * ldbt + kt + 32 + skof, 0, 1);
        }

        // --- A fragments: ISA 16-bit 16x32 layout (lane&15 = row,
        //     lanes>=16 start at K=8; elems 0..7 = K base.., 8..15 = K base+16..)
        const int la = lane & 15;
        const int ka = (lane >> 4) * 8;
        v16h afrag[4];
        for (int i = 0; i < 4; i++) {
            const f16* pa = As + (wm * 64 + i * 16 + la) * LDS_STRIDE + ka;
            h8 x0 = *(const h8*)pa;
            h8 x1 = *(const h8*)(pa + 16);
            v16h f;
            for (int j = 0; j < 8; j++) { f[j] = x0[j]; f[j + 8] = x1[j]; }
            afrag[i] = f;
        }
        // --- B fragments: same K layout as A (lanes>=16 start at K=8; elems 8..15 = K base+16..)
        const int cbn = lane & 15;
        const int kb  = (lane >> 4) * 8;
        v16h bfrag[2];
        for (int j = 0; j < 2; j++) {
            const f16* pb = Bs + (wn * 32 + j * 16 + cbn) * LDS_STRIDE + kb;
            h8 x0 = *(const h8*)pb;
            h8 x1 = *(const h8*)(pb + 16);
            v16h f;
            for (int q = 0; q < 8; q++) { f[q] = x0[q]; f[q + 8] = x1[q]; }
            bfrag[j] = f;
        }

        for (int i = 0; i < 4; i++)
            for (int j = 0; j < 2; j++)
                acc[i][j] = __builtin_amdgcn_wmma_f32_16x16x32_f16(
                    false, afrag[i], false, bfrag[j], (short)0, acc[i][j], false, false);
    }

    // --- epilogue: C layout VGPR v -> M=v (lanes 0-15) / M=v+8 (lanes 16-31)
    const int rlo = (lane >> 4) * 8;
    const int cc  = lane & 15;
    for (int i = 0; i < 4; i++) {
        for (int j = 0; j < 2; j++) {
            int rb    = m0 + wm * 64 + i * 16 + rlo;
            int cbase = n0 + wn * 32 + j * 16 + cc;
            float bv  = bias ? bias[cbase] : 0.0f;
            for (int v = 0; v < 8; v++) {
                int r = rb + v;
                if (r < M) {
                    float x = acc[i][j][v] + bv;
                    if (act == 1)      x = 0.5f * x * (1.0f + erff(x * 0.70710678118654752f));
                    else if (act == 2) x = tanhf(x);
                    if (C)   C[(size_t)r * ldc + cbase]     = x;
                    if (C16) C16[(size_t)r * ldc16 + cbase] = (f16)x;
                }
            }
        }
    }
}

// ---------------------------------------------------------------------------
// Weight preparation: emit f16 in B-transposed layout BT[Nc][Kpad]
// ---------------------------------------------------------------------------
__global__ void k_wBT(f16* __restrict__ dst, const float* __restrict__ src,
                      int K, int Nc, int Kpad)   // src [K,Nc] -> dst[n*Kpad + k]
{
    int idx = blockIdx.x * blockDim.x + threadIdx.x;
    if (idx >= Nc * Kpad) return;
    int n = idx / Kpad, k = idx - n * Kpad;
    dst[idx] = (k < K) ? (f16)src[(size_t)k * Nc + n] : (f16)0.0f;
}

__global__ void k_wcast(f16* __restrict__ dst, const float* __restrict__ src, int n)
{   // src already [Nout,K] row-major (e.g. GRU weights) -> straight cast
    int idx = blockIdx.x * blockDim.x + threadIdx.x;
    if (idx < n) dst[idx] = (f16)src[idx];
}

__global__ void k_w_headsT(f16* __restrict__ dst, const float* __restrict__ src)
{   // src (4,512,128) -> BT[512][512], dst[(h*128+j)*512 + k] = src[h][k][j]
    int idx = blockIdx.x * blockDim.x + threadIdx.x;
    if (idx >= 512 * 512) return;
    int n = idx >> 9, k = idx & 511, h = n >> 7, j = n & 127;
    dst[idx] = (f16)src[((size_t)h * 512 + k) * 128 + j];
}

// ---------------------------------------------------------------------------
// Elementwise / index kernels (closed-form complete binary tree structure)
// ---------------------------------------------------------------------------
__global__ void k_edge_attr(const int* __restrict__ pc, const int* __restrict__ cc,
                            const int* __restrict__ bt, const float* __restrict__ W,
                            const float* __restrict__ b, float* __restrict__ out)
{
    int idx = blockIdx.x * blockDim.x + threadIdx.x;
    if (idx >= EE * 24) return;
    int e = idx / 24, c = idx - e * 24;
    int p = min(max(pc[e], 0), 6), q = min(max(cc[e], 0), 6), t = min(max(bt[e], 0), 2);
    out[idx] = W[p * 24 + c] + W[(7 + q) * 24 + c] + W[(14 + t) * 24 + c] + b[c];
}

__global__ void k_cast_f16(f16* __restrict__ dst, int dld,
                           const float* __restrict__ src, int sld, int rows, int cols)
{
    int idx = blockIdx.x * blockDim.x + threadIdx.x;
    if (idx >= rows * cols) return;
    int r = idx / cols, c = idx - r * cols;
    dst[(size_t)r * dld + c] = (f16)src[(size_t)r * sld + c];
}

__global__ void k_build_cm(const f16* __restrict__ h16, const float* __restrict__ ea,
                           f16* __restrict__ out)   // [E,544] = [child_h | edge_attr | pad]
{
    int idx = blockIdx.x * blockDim.x + threadIdx.x;
    if (idx >= EE * 544) return;
    int e = idx / 544, c = idx - e * 544;
    int g = e / 62, j = e - g * 62;
    int child = g * 63 + j + 1;
    f16 v = (f16)0.0f;
    if (c < 512)      v = h16[(size_t)child * 512 + c];
    else if (c < 536) v = (f16)ea[(size_t)e * 24 + (c - 512)];
    out[idx] = v;
}

__global__ void k_build_at(const f16* __restrict__ h16, f16* __restrict__ out) // [E,1024]
{
    int idx = blockIdx.x * blockDim.x + threadIdx.x;
    if (idx >= EE * 1024) return;
    int e = idx >> 10, c = idx & 1023;
    int g = e / 62, j = e - g * 62;
    int child = g * 63 + j + 1;
    int par   = g * 63 + (j >> 1);
    out[idx] = (c < 512) ? h16[(size_t)child * 512 + c]
                         : h16[(size_t)par * 512 + (c - 512)];
}

__global__ void k_attn_pair(const float* __restrict__ s1mid, const float* __restrict__ w2,
                            const float* __restrict__ b2, float* __restrict__ aw, int pairs)
{
    int t = blockIdx.x * blockDim.x + threadIdx.x;
    if (t >= pairs) return;
    const float* p0 = s1mid + (size_t)(2 * t) * 128;
    const float* p1 = p0 + 128;
    float s0 = b2[0], s1 = b2[0];
    for (int k = 0; k < 128; k++) { s0 += p0[k] * w2[k]; s1 += p1[k] * w2[k]; }
    float m = fmaxf(s0, s1);
    float e0 = expf(s0 - m), e1 = expf(s1 - m);
    float inv = 1.0f / (e0 + e1);
    aw[2 * t] = e0 * inv; aw[2 * t + 1] = e1 * inv;
}

__global__ void k_child_agg(const float* __restrict__ aw, const float* __restrict__ msgs,
                            float* __restrict__ out)   // [N,512]
{
    int idx = blockIdx.x * blockDim.x + threadIdx.x;
    if (idx >= NN * 512) return;
    int n = idx >> 9, c = idx & 511;
    int g = n / 63, i = n - g * 63;
    float r = 0.0f;
    if (i <= 30) {
        int e0 = g * 62 + 2 * i;
        r = aw[e0] * msgs[(size_t)e0 * 512 + c] + aw[e0 + 1] * msgs[(size_t)(e0 + 1) * 512 + c];
    }
    out[idx] = r;
}

__global__ void k_gather_sib(const f16* __restrict__ h16, f16* __restrict__ out)
{
    int idx = blockIdx.x * blockDim.x + threadIdx.x;
    if (idx >= NN * 512) return;
    int n = idx >> 9, c = idx & 511;
    int g = n / 63, i = n - g * 63;
    f16 v = (f16)0.0f;
    if (i > 0) {
        int s = (i & 1) ? i + 1 : i - 1;
        v = h16[(size_t)(g * 63 + s) * 512 + c];
    }
    out[idx] = v;
}

__global__ __launch_bounds__(256)
void k_gru_ln(const float* __restrict__ gi, const float* __restrict__ gh,
              float* __restrict__ h, const float* __restrict__ g,
              const float* __restrict__ b, int rows)
{
    int wave = threadIdx.x >> 5, lane = threadIdx.x & 31;
    int row = blockIdx.x * 8 + wave;
    if (row >= rows) return;
    const float* pi = gi + (size_t)row * 1536;
    const float* ph = gh + (size_t)row * 1536;
    float* hr = h + (size_t)row * 512;
    float v[16], s = 0.0f, sq = 0.0f;
    for (int jj = 0; jj < 16; jj++) {
        int c = lane + jj * 32;
        float r  = 1.0f / (1.0f + expf(-(pi[c] + ph[c])));
        float z  = 1.0f / (1.0f + expf(-(pi[c + 512] + ph[c + 512])));
        float n_ = tanhf(pi[c + 1024] + r * ph[c + 1024]);
        float val = (1.0f - z) * n_ + z * hr[c];
        v[jj] = val; s += val; sq += val * val;
    }
    for (int o = 16; o >= 1; o >>= 1) { s += __shfl_xor(s, o, 32); sq += __shfl_xor(sq, o, 32); }
    float mu = s * (1.0f / 512.0f);
    float var = sq * (1.0f / 512.0f) - mu * mu;
    float rs = rsqrtf(var + 1e-5f);
    for (int jj = 0; jj < 16; jj++) {
        int c = lane + jj * 32;
        hr[c] = (v[jj] - mu) * rs * g[c] + b[c];
    }
}

__global__ __launch_bounds__(256)
void k_ln_rows(const float* __restrict__ x, const float* __restrict__ g,
               const float* __restrict__ b, float* __restrict__ out, int rows)
{
    int wave = threadIdx.x >> 5, lane = threadIdx.x & 31;
    int row = blockIdx.x * 8 + wave;
    if (row >= rows) return;
    const float* xr = x + (size_t)row * 512;
    float* orow = out + (size_t)row * 512;
    float v[16], s = 0.0f, sq = 0.0f;
    for (int jj = 0; jj < 16; jj++) {
        float val = xr[lane + jj * 32];
        v[jj] = val; s += val; sq += val * val;
    }
    for (int o = 16; o >= 1; o >>= 1) { s += __shfl_xor(s, o, 32); sq += __shfl_xor(sq, o, 32); }
    float mu = s * (1.0f / 512.0f);
    float var = sq * (1.0f / 512.0f) - mu * mu;
    float rs = rsqrtf(var + 1e-5f);
    for (int jj = 0; jj < 16; jj++) {
        int c = lane + jj * 32;
        orow[c] = (v[jj] - mu) * rs * g[c] + b[c];
    }
}

__global__ void k_build_td(const f16* __restrict__ h16, const float* __restrict__ href,
                           f16* __restrict__ out, int t, int cnt)
{
    int idx = blockIdx.x * blockDim.x + threadIdx.x;
    if (idx >= cnt * 1024) return;
    int r = idx >> 10, c = idx & 1023;
    int g = r >> t, i = r & ((1 << t) - 1);
    int li = (1 << t) - 1 + i;
    int node = g * 63 + li;
    f16 v;
    if (c < 512) v = h16[(size_t)node * 512 + c];
    else if (t == 0) v = (f16)0.0f;
    else {
        int par = g * 63 + ((li - 1) >> 1);
        v = (f16)href[(size_t)par * 512 + (c - 512)];
    }
    out[idx] = v;
}

__global__ void k_scatter_td(const float* __restrict__ src, float* __restrict__ href,
                             int t, int cnt)
{
    int idx = blockIdx.x * blockDim.x + threadIdx.x;
    if (idx >= cnt * 512) return;
    int r = idx >> 9, c = idx & 511;
    int g = r >> t, i = r & ((1 << t) - 1);
    int node = g * 63 + (1 << t) - 1 + i;
    href[(size_t)node * 512 + c] = src[idx];
}

__global__ void k_scores4(const float* __restrict__ s1, const float* __restrict__ W2,
                          const float* __restrict__ b2, float* __restrict__ out)
{
    int idx = blockIdx.x * blockDim.x + threadIdx.x;
    if (idx >= NN * 4) return;
    int node = idx >> 2, h = idx & 3;
    const float* p = s1 + (size_t)node * 512 + h * 128;
    const float* w = W2 + h * 128;
    float acc = b2[h];
    for (int k = 0; k < 128; k++) acc += p[k] * w[k];
    out[idx] = acc;
}

__global__ __launch_bounds__(256)
void k_pool(const float* __restrict__ scores, const float* __restrict__ href,
            f16* __restrict__ pooled16, float* __restrict__ hbranch)
{
    __shared__ float sA[63 * 4];
    __shared__ float sm[4], sden[4];
    int g = blockIdx.x, t = threadIdx.x;
    if (t < 252) sA[t] = scores[(size_t)(g * 63 + (t >> 2)) * 4 + (t & 3)];
    __syncthreads();
    if (t < 4) {
        float m = -1e30f;
        for (int i = 0; i < 63; i++) m = fmaxf(m, sA[i * 4 + t]);
        float s = 0.0f;
        for (int i = 0; i < 63; i++) s += expf(sA[i * 4 + t] - m);
        sm[t] = m; sden[t] = s;
    }
    __syncthreads();
    if (t < 252) sA[t] = expf(sA[t] - sm[t & 3]) / sden[t & 3];
    __syncthreads();
    for (int o = t; o < 2048; o += 256) {
        int h = o >> 9, dim = o & 511;
        float acc = 0.0f;
        for (int i = 0; i < 63; i++)
            acc += sA[i * 4 + h] * href[(size_t)(g * 63 + i) * 512 + dim];
        pooled16[(size_t)g * 2048 + o] = (f16)acc;
    }
    for (int dim = t; dim < 512; dim += 256) {
        float acc = 0.0f;
        for (int i = 0; i < 31; i++) acc += href[(size_t)(g * 63 + i) * 512 + dim];
        hbranch[(size_t)g * 512 + dim] = acc / (31.0f + 1e-8f);
    }
}

__global__ void k_build_fused(const float* __restrict__ hg, const float* __restrict__ hb,
                              const float* __restrict__ de_emb, const int* __restrict__ depth,
                              f16* __restrict__ out)   // [G,1056]
{
    int idx = blockIdx.x * blockDim.x + threadIdx.x;
    if (idx >= GG * 1056) return;
    int g = idx / 1056, c = idx - g * 1056;
    f16 v = (f16)0.0f;
    if (c < 512)       v = (f16)hg[(size_t)g * 512 + c];
    else if (c < 1024) v = (f16)hb[(size_t)g * 512 + (c - 512)];
    else if (c < 1032) {
        int md = min(max(depth[g * 63 + 62], 0), 31);  // max heap depth per graph
        v = (f16)de_emb[md * 8 + (c - 1024)];
    }
    out[idx] = v;
}

// ---------------------------------------------------------------------------
extern "C" void kernel_launch(void* const* d_in, const int* in_sizes, int n_in,
                              void* d_out, int out_size, void* d_ws, size_t ws_size,
                              hipStream_t stream) {
    (void)in_sizes; (void)n_in; (void)out_size; (void)ws_size;
    const float* in_h    = (const float*)d_in[0];
    const int*   depth   = (const int*)d_in[3];
    const int*   pc      = (const int*)d_in[6];
    const int*   cc      = (const int*)d_in[7];
    const int*   bt      = (const int*)d_in[8];
    const float* link_W  = (const float*)d_in[10];
    const float* link_b  = (const float*)d_in[11];
    const float* cm_W1   = (const float*)d_in[12];
    const float* cm_b1   = (const float*)d_in[13];
    const float* cm_W2   = (const float*)d_in[14];
    const float* cm_b2   = (const float*)d_in[15];
    const float* at_W1   = (const float*)d_in[16];
    const float* at_b1   = (const float*)d_in[17];
    const float* at_W2   = (const float*)d_in[18];
    const float* at_b2   = (const float*)d_in[19];
    const float* sb_W1   = (const float*)d_in[20];
    const float* sb_b1   = (const float*)d_in[21];
    const float* sb_W2   = (const float*)d_in[22];
    const float* sb_b2   = (const float*)d_in[23];
    const float* gru_Wih = (const float*)d_in[24];
    const float* gru_Whh = (const float*)d_in[25];
    const float* gru_bih = (const float*)d_in[26];
    const float* gru_bhh = (const float*)d_in[27];
    const float* ln_g    = (const float*)d_in[28];
    const float* ln_b    = (const float*)d_in[29];
    const float* td_W1   = (const float*)d_in[30];
    const float* td_b1   = (const float*)d_in[31];
    const float* td_W2   = (const float*)d_in[32];
    const float* td_b2   = (const float*)d_in[33];
    const float* td_g    = (const float*)d_in[34];
    const float* td_bb   = (const float*)d_in[35];
    const float* ah_W1   = (const float*)d_in[36];
    const float* ah_b1   = (const float*)d_in[37];
    const float* ah_W2   = (const float*)d_in[38];
    const float* ah_b2   = (const float*)d_in[39];
    const float* hp_W    = (const float*)d_in[40];
    const float* hp_b    = (const float*)d_in[41];
    const float* de_emb  = (const float*)d_in[42];
    const float* fu_W1   = (const float*)d_in[43];
    const float* fu_b1   = (const float*)d_in[44];
    const float* fu_W2   = (const float*)d_in[45];
    const float* fu_b2   = (const float*)d_in[46];
    const float* po_g    = (const float*)d_in[47];
    const float* po_b    = (const float*)d_in[48];

    // ---- workspace carve (256B aligned) ----
    char* wsp = (char*)d_ws;
    auto carve = [&](size_t bytes) -> void* {
        void* r = (void*)wsp;
        wsp += (bytes + 255) & ~(size_t)255;
        return r;
    };
    float* h_cur   = (float*)carve((size_t)NN * 512 * 4);
    f16*   h16     = (f16*)  carve((size_t)NN * 512 * 2);
    f16*   sib16   = (f16*)  carve((size_t)NN * 512 * 2);
    f16*   big16   = (f16*)  carve((size_t)NN * 1024 * 2);  // Xcat / at_in / x16 / td_in / fuse_in
    f16*   mid16   = (f16*)  carve((size_t)NN * 512 * 2);
    float* eattr   = (float*)carve((size_t)EE * 24 * 4);
    float* msgs    = (float*)carve((size_t)EE * 512 * 4);
    float* cagg    = (float*)carve((size_t)NN * 512 * 4);   // also s1 / tdout
    float* gi      = (float*)carve((size_t)NN * 1536 * 4);
    float* gh      = (float*)carve((size_t)NN * 1536 * 4);
    float* s1mid   = (float*)carve((size_t)EE * 128 * 4);   // also scores [N,4]
    float* aw      = (float*)carve((size_t)EE * 4);
    float* h_ref   = (float*)carve((size_t)NN * 512 * 4);
    f16*   pooled16= (f16*)  carve((size_t)GG * 2048 * 2);
    float* hbranch = (float*)carve((size_t)GG * 512 * 4);
    float* hglobal = (float*)carve((size_t)GG * 512 * 4);
    float* gtmp    = (float*)carve((size_t)GG * 512 * 4);
    f16* wcm1 = (f16*)carve((size_t)512 * 544 * 2);
    f16* wcm2 = (f16*)carve((size_t)512 * 512 * 2);
    f16* wat1 = (f16*)carve((size_t)128 * 1024 * 2);
    f16* wsb1 = (f16*)carve((size_t)512 * 512 * 2);
    f16* wsb2 = (f16*)carve((size_t)512 * 512 * 2);
    f16* wih  = (f16*)carve((size_t)1536 * 1024 * 2);
    f16* whh  = (f16*)carve((size_t)1536 * 512 * 2);
    f16* wtd1 = (f16*)carve((size_t)512 * 1024 * 2);
    f16* wtd2 = (f16*)carve((size_t)512 * 512 * 2);
    f16* wah1 = (f16*)carve((size_t)512 * 512 * 2);
    f16* whp  = (f16*)carve((size_t)512 * 2048 * 2);
    f16* wfu1 = (f16*)carve((size_t)512 * 1056 * 2);
    f16* wfu2 = (f16*)carve((size_t)512 * 512 * 2);

    auto blocks = [](long long n) { return (unsigned)((n + 255) / 256); };
    auto gemm = [&](const f16* A, int lda, const f16* BT, int ldbt, const float* bias,
                    float* C, int ldc, f16* C16, int ldc16, int M, int Nc, int K, int act) {
        dim3 grid(Nc / 128, (M + 127) / 128);
        wmma_gemm<<<grid, 256, 0, stream>>>(A, lda, BT, ldbt, bias, C, ldc, C16, ldc16, M, Nc, K, act);
    };

    hipMemcpyAsync(h_cur, in_h, (size_t)NN * 512 * 4, hipMemcpyDeviceToDevice, stream);
    k_edge_attr<<<blocks((long long)EE * 24), 256, 0, stream>>>(pc, cc, bt, link_W, link_b, eattr);

    // ================= bottom-up Tree-MPNN layers =================
    for (int l = 0; l < 3; l++) {
        // weight prep: f16 in BT[N][Kpad] layout (GRU weights are already [Nout,K])
        k_wBT<<<blocks(512 * 544), 256, 0, stream>>>(wcm1, cm_W1 + (size_t)l * 536 * 512, 536, 512, 544);
        k_wBT<<<blocks(512 * 512), 256, 0, stream>>>(wcm2, cm_W2 + (size_t)l * 512 * 512, 512, 512, 512);
        k_wBT<<<blocks(128 * 1024), 256, 0, stream>>>(wat1, at_W1 + (size_t)l * 1024 * 128, 1024, 128, 1024);
        k_wBT<<<blocks(512 * 512), 256, 0, stream>>>(wsb1, sb_W1 + (size_t)l * 512 * 512, 512, 512, 512);
        k_wBT<<<blocks(512 * 512), 256, 0, stream>>>(wsb2, sb_W2 + (size_t)l * 512 * 512, 512, 512, 512);
        k_wcast<<<blocks(1536 * 1024), 256, 0, stream>>>(wih, gru_Wih + (size_t)l * 1536 * 1024, 1536 * 1024);
        k_wcast<<<blocks(1536 * 512), 256, 0, stream>>>(whh, gru_Whh + (size_t)l * 1536 * 512, 1536 * 512);

        k_cast_f16<<<blocks((long long)NN * 512), 256, 0, stream>>>(h16, 512, h_cur, 512, NN, 512);

        // child message MLP: gelu([child_h|edge]@W1+b1)@W2+b2 -> msgs f32
        k_build_cm<<<blocks((long long)EE * 544), 256, 0, stream>>>(h16, eattr, big16);
        gemm(big16, 544, wcm1, 544, cm_b1 + l * 512, nullptr, 0, mid16, 512, EE, 512, 544, 1);
        gemm(mid16, 512, wcm2, 512, cm_b2 + l * 512, msgs, 512, nullptr, 0, EE, 512, 512, 0);

        // attention over 2 children per parent
        k_build_at<<<blocks((long long)EE * 1024), 256, 0, stream>>>(h16, big16);
        gemm(big16, 1024, wat1, 1024, at_b1 + l * 128, s1mid, 128, nullptr, 0, EE, 128, 1024, 2);
        k_attn_pair<<<blocks(EE / 2), 256, 0, stream>>>(s1mid, at_W2 + l * 128, at_b2 + l, aw, EE / 2);
        k_child_agg<<<blocks((long long)NN * 512), 256, 0, stream>>>(aw, msgs, cagg);

        // sibling MLP (sibling gather is a row permutation of h)
        k_gather_sib<<<blocks((long long)NN * 512), 256, 0, stream>>>(h16, sib16);
        gemm(sib16, 512, wsb1, 512, sb_b1 + l * 512, nullptr, 0, mid16, 512, NN, 512, 512, 1);
        // write sib_feat (f16) directly into columns [512,1024) of x16 = big16
        gemm(mid16, 512, wsb2, 512, sb_b2 + l * 512, nullptr, 0, big16 + 512, 1024, NN, 512, 512, 0);
        // write child_agg (f16) into columns [0,512) of x16
        k_cast_f16<<<blocks((long long)NN * 512), 256, 0, stream>>>(big16, 1024, cagg, 512, NN, 512);

        // GRU gates + LayerNorm
        gemm(big16, 1024, wih, 1024, gru_bih + l * 1536, gi, 1536, nullptr, 0, NN, 1536, 1024, 0);
        gemm(h16, 512, whh, 512, gru_bhh + l * 1536, gh, 1536, nullptr, 0, NN, 1536, 512, 0);
        k_gru_ln<<<(NN + 7) / 8, 256, 0, stream>>>(gi, gh, h_cur, ln_g + l * 512, ln_b + l * 512, NN);
    }

    // ================= top-down refinement (level-compact) =================
    k_wBT<<<blocks(512 * 1024), 256, 0, stream>>>(wtd1, td_W1, 1024, 512, 1024);
    k_wBT<<<blocks(512 * 512), 256, 0, stream>>>(wtd2, td_W2, 512, 512, 512);
    k_w_headsT<<<blocks(512 * 512), 256, 0, stream>>>(wah1, ah_W1);
    k_wBT<<<blocks(512 * 2048), 256, 0, stream>>>(whp, hp_W, 2048, 512, 2048);
    k_wBT<<<blocks(512 * 1056), 256, 0, stream>>>(wfu1, fu_W1, 1032, 512, 1056);
    k_wBT<<<blocks(512 * 512), 256, 0, stream>>>(wfu2, fu_W2, 512, 512, 512);

    k_cast_f16<<<blocks((long long)NN * 512), 256, 0, stream>>>(h16, 512, h_cur, 512, NN, 512);
    for (int t = 0; t < 6; t++) {
        int cnt = GG << t;   // nodes at heap depth t (contiguous per graph)
        k_build_td<<<blocks((long long)cnt * 1024), 256, 0, stream>>>(h16, h_ref, big16, t, cnt);
        gemm(big16, 1024, wtd1, 1024, td_b1, nullptr, 0, mid16, 512, cnt, 512, 1024, 1);
        gemm(mid16, 512, wtd2, 512, td_b2, cagg, 512, nullptr, 0, cnt, 512, 512, 0);
        k_scatter_td<<<blocks((long long)cnt * 512), 256, 0, stream>>>(cagg, h_ref, t, cnt);
    }
    k_ln_rows<<<(NN + 7) / 8, 256, 0, stream>>>(h_ref, td_g, td_bb, h_ref, NN);

    // ================= branching-aware pooling =================
    k_cast_f16<<<blocks((long long)NN * 512), 256, 0, stream>>>(h16, 512, h_ref, 512, NN, 512);
    gemm(h16, 512, wah1, 512, ah_b1, cagg, 512, nullptr, 0, NN, 512, 512, 2);   // tanh, 4 heads fused
    k_scores4<<<blocks((long long)NN * 4), 256, 0, stream>>>(cagg, ah_W2, ah_b2, s1mid);
    k_pool<<<GG, 256, 0, stream>>>(s1mid, h_ref, pooled16, hbranch);
    gemm(pooled16, 2048, whp, 2048, hp_b, hglobal, 512, nullptr, 0, GG, 512, 2048, 0);

    // ================= fusion head =================
    k_build_fused<<<blocks((long long)GG * 1056), 256, 0, stream>>>(hglobal, hbranch, de_emb, depth, big16);
    gemm(big16, 1056, wfu1, 1056, fu_b1, nullptr, 0, mid16, 512, GG, 512, 1056, 1);
    gemm(mid16, 512, wfu2, 512, fu_b2, gtmp, 512, nullptr, 0, GG, 512, 512, 0);
    k_ln_rows<<<(GG + 7) / 8, 256, 0, stream>>>(gtmp, po_g, po_b, (float*)d_out, GG);
}
